// MultiheadAttention_10995116277871
// MI455X (gfx1250) — matrix operand order repaired and checked by reading, not compiled, not run on hardware
//
#include <hip/hip_runtime.h>
#include <hip/hip_bf16.h>

// ---------------------------------------------------------------------------
// MultiheadAttention forward for MI455X (gfx1250), bf16 WMMA + flash softmax
//   S=2048, B=2, E=1024, H=16, hd=64
// Register-blocked v2: fat accumulator tiles to raise WMMA : VMEM issue ratio.
// ---------------------------------------------------------------------------

typedef __bf16 bf16;
typedef __attribute__((ext_vector_type(16))) __bf16 bf16x16;
typedef __attribute__((ext_vector_type(8)))  __bf16 bf16x8;
typedef __attribute__((ext_vector_type(8)))  float  f32x8;

#define S_LEN   2048
#define BATCH   2
#define EMB     1024
#define HEADS   16
#define HD      64
#define BHEADS  (BATCH * HEADS)        // 32
#define M_ROWS  (S_LEN * BATCH)        // 4096
#define LOG2E   1.44269504088896f
#define Q_SCALE 0.125f                 // hd^-0.5 = 1/8

// ---------------- WMMA helpers --------------------------------------------

__device__ __forceinline__ f32x8 wmma_bf16(bf16x16 a, bf16x16 b, f32x8 c) {
  // D = A(16x32) * B(32x16) + C, f32 accumulate
  return __builtin_amdgcn_wmma_f32_16x16x32_bf16(
      /*neg_a=*/false, a, /*neg_b=*/false, b,
      /*c_mod=*/(short)0, c, /*reuse_a=*/false, /*reuse_b=*/false);
}

// A operand (16x32, 16-bit): lane holds row M = rowBase + lane%16.
// elements 0..7  -> K = k + (lane<16 ? 0 : 8)  + i
// elements 8..15 -> K = k + (lane<16 ? 16 : 24) + (i-8)
__device__ __forceinline__ bf16x16
load_a_tile(const bf16* __restrict__ base, int rowStride, int rowBase, int k, int lane) {
  const bf16* p = base + (size_t)(rowBase + (lane & 15)) * rowStride
                       + k + ((lane & 16) ? 8 : 0);
  union { bf16x16 v; bf16x8 h[2]; } u;
  u.h[0] = *(const bf16x8*)(p);
  u.h[1] = *(const bf16x8*)(p + 16);
  return u.v;
}

// B operand (32x16, 16-bit, KxN): lane holds column N = colBase + lane%16,
// elements 0..7  -> K = k + (lane<16 ? 0 : 8)  + i
// elements 8..15 -> K = k + (lane<16 ? 16 : 24) + (i-8)
__device__ __forceinline__ bf16x16
load_b_tile(const bf16* __restrict__ base, int rowStride, int colBase, int k, int lane) {
  const bf16* p = base + (size_t)(colBase + (lane & 15)) * rowStride
                       + k + ((lane & 16) ? 8 : 0);
  union { bf16x16 v; bf16x8 h[2]; } u;
  u.h[0] = *(const bf16x8*)(p);
  u.h[1] = *(const bf16x8*)(p + 16);
  return u.v;
}

// ---------------- Stage 1: f32 -> bf16 cast -------------------------------

__global__ void cast_f32_to_bf16(const float* __restrict__ in,
                                 bf16* __restrict__ out, int n) {
  int i = blockIdx.x * blockDim.x + threadIdx.x;
  if (i < n) out[i] = (bf16)in[i];
}

// ---------------- Stage 2: fused QKV projection ---------------------------
// x[M,E] (bf16) @ {wq,wk,wv}[N,K]^T + bias -> head-split bf16 tensors:
//   qh, kh : [B*H, S, hd]   (q pre-scaled by hd^-0.5)
//   vT     : [B*H, hd, S]   (transposed for PV B-operand)
// Each wave owns a 64x16 output block for all three matrices:
// per k-step 4 A + 3 B operand loads feed 12 WMMAs.
#define PM 4
__global__ __launch_bounds__(256)
void qkv_proj_kernel(const bf16* __restrict__ xb,
                     const bf16* __restrict__ wqb,
                     const bf16* __restrict__ wkb,
                     const bf16* __restrict__ wvb,
                     const float* __restrict__ bq,
                     const float* __restrict__ bk,
                     const float* __restrict__ bv,
                     bf16* __restrict__ qh,
                     bf16* __restrict__ kh,
                     bf16* __restrict__ vT) {
  const int lane = threadIdx.x & 31;
  const int wave = threadIdx.x >> 5;
  const int tile = blockIdx.x * 8 + wave;     // (M/64) * (E/16) = 4096 tiles
  const int mt = tile / (EMB / 16);
  const int nt = tile % (EMB / 16);
  const int m0 = mt * (16 * PM), n0 = nt * 16;

  f32x8 cq[PM] = {}, ck[PM] = {}, cv[PM] = {};

  for (int k = 0; k < EMB; k += 32) {
    bf16x16 a[PM];
#pragma unroll
    for (int i = 0; i < PM; ++i)
      a[i] = load_a_tile(xb, EMB, m0 + 16 * i, k, lane);

    bf16x16 b;
    b = load_b_tile(wqb, EMB, n0, k, lane);
#pragma unroll
    for (int i = 0; i < PM; ++i) cq[i] = wmma_bf16(a[i], b, cq[i]);
    b = load_b_tile(wkb, EMB, n0, k, lane);
#pragma unroll
    for (int i = 0; i < PM; ++i) ck[i] = wmma_bf16(a[i], b, ck[i]);
    b = load_b_tile(wvb, EMB, n0, k, lane);
#pragma unroll
    for (int i = 0; i < PM; ++i) cv[i] = wmma_bf16(a[i], b, cv[i]);
  }

  const int n = n0 + (lane & 15);             // output feature
  const int h = n >> 6, d = n & 63;           // head / dim-in-head
  const float bqv = bq[n], bkv = bk[n], bvv = bv[n];

#pragma unroll
  for (int i = 0; i < PM; ++i) {
#pragma unroll
    for (int r = 0; r < 8; ++r) {
      const int m = m0 + 16 * i + r + ((lane & 16) ? 8 : 0);  // row = s*B + b
      const int s = m / BATCH, b = m % BATCH;
      const size_t hb = (size_t)(b * HEADS + h);
      qh[(hb * S_LEN + s) * HD + d] = (bf16)((cq[i][r] + bqv) * Q_SCALE);
      kh[(hb * S_LEN + s) * HD + d] = (bf16)(ck[i][r] + bkv);
      vT[(hb * HD + d) * S_LEN + s] = (bf16)(cv[i][r] + bvv);
    }
  }
}

// ---------------- Stage 3: flash attention --------------------------------
// One wave owns QBLK (head, 16-query-row) tiles. Computes S^T = K * Q^T so
// the score D-layout (lane = q column, vgpr = kpos row) packs in-lane into
// the 16-bit A-operand layout of the P*V WMMA — no transpose needed.
// K A-operands and V B-operands are loaded once per step and reused by both
// query tiles: 8 operand loads per 16 WMMAs.
#define QBLK 2
__global__ __launch_bounds__(256)
void flash_attn_kernel(const bf16* __restrict__ qh,
                       const bf16* __restrict__ kh,
                       const bf16* __restrict__ vT,
                       bf16* __restrict__ ctx /* [M_ROWS, EMB] */) {
  const int lane = threadIdx.x & 31;
  const int wave = threadIdx.x >> 5;
  const int n    = blockIdx.x;                     // head-batch index 0..31
  const int qt0  = (blockIdx.y * 8 + wave) * QBLK; // first q tile of this wave

  const bf16* qb = qh + (size_t)n * S_LEN * HD;
  const bf16* kb = kh + (size_t)n * S_LEN * HD;
  const bf16* vb = vT + (size_t)n * HD * S_LEN;

  bf16x16 qB0[QBLK], qB1[QBLK];                    // Q^T, d-chunks [0,32),[32,64)
  f32x8 o[QBLK][4] = {};
  float mrun[QBLK], lrun[QBLK];
#pragma unroll
  for (int j = 0; j < QBLK; ++j) {
    const int q0 = (qt0 + j) * 16;
    qB0[j] = load_b_tile(qb, HD, q0, 0, lane);
    qB1[j] = load_b_tile(qb, HD, q0, 32, lane);
    mrun[j] = -1e30f;
    lrun[j] = 0.0f;
  }

  for (int t = 0; t < S_LEN / 32; ++t) {
    const int k0 = t * 32;

    if (t + 1 < S_LEN / 32) {                      // warm caches for next step
      __builtin_prefetch(kb + (size_t)(k0 + 32 + (lane & 15)) * HD, 0, 1);
      __builtin_prefetch(vb + (size_t)(lane & 15) * S_LEN + k0 + 32, 0, 1);
    }

    // Shared operands for this k-step.
    bf16x16 kA[4];                                 // [kpos block][d chunk]
    kA[0] = load_a_tile(kb, HD, k0,      0, lane);
    kA[1] = load_a_tile(kb, HD, k0,     32, lane);
    kA[2] = load_a_tile(kb, HD, k0 + 16,  0, lane);
    kA[3] = load_a_tile(kb, HD, k0 + 16, 32, lane);
    bf16x16 vB[4];                                 // V^T columns d-tile 0..3
#pragma unroll
    for (int dt = 0; dt < 4; ++dt)
      vB[dt] = load_b_tile(vb, S_LEN, dt * 16, k0, lane);

#pragma unroll
    for (int j = 0; j < QBLK; ++j) {
      // S^T tiles: rows = kpos, cols = q. A = K rows, B = Q^T.
      f32x8 s0 = {}, s1 = {};
      s0 = wmma_bf16(kA[0], qB0[j], s0);
      s0 = wmma_bf16(kA[1], qB1[j], s0);
      s1 = wmma_bf16(kA[2], qB0[j], s1);
      s1 = wmma_bf16(kA[3], qB1[j], s1);

      // Each lane holds 16 scores of query row q = (qt0+j)*16 + (lane&15):
      //   s0[r] -> kpos = k0 +      r + (lane<16 ? 0 : 8)
      //   s1[r] -> kpos = k0 + 16 + r + (lane<16 ? 0 : 8)
      float mloc = s0[0];
#pragma unroll
      for (int r = 1; r < 8; ++r) mloc = fmaxf(mloc, s0[r]);
#pragma unroll
      for (int r = 0; r < 8; ++r) mloc = fmaxf(mloc, s1[r]);
      const float mstep = fmaxf(mloc, __shfl_xor(mloc, 16, 32));
      const float mnew  = fmaxf(mrun[j], mstep);
      const float alpha = exp2f((mrun[j] - mnew) * LOG2E);

      float pv[16];
      float lloc = 0.0f;
#pragma unroll
      for (int r = 0; r < 8; ++r) { pv[r]     = exp2f((s0[r] - mnew) * LOG2E); lloc += pv[r]; }
#pragma unroll
      for (int r = 0; r < 8; ++r) { pv[8 + r] = exp2f((s1[r] - mnew) * LOG2E); lloc += pv[8 + r]; }
      lrun[j] = lrun[j] * alpha + lloc + __shfl_xor(lloc, 16, 32);
      mrun[j] = mnew;

      // Rescale O: vgpr r is row q = r + (lane<16?0:8); alpha for row q lives
      // in lane q (and q+16).
#pragma unroll
      for (int r = 0; r < 8; ++r) {
        const float ar = __shfl(alpha, r + ((lane & 16) ? 8 : 0), 32);
        o[j][0][r] *= ar; o[j][1][r] *= ar; o[j][2][r] *= ar; o[j][3][r] *= ar;
      }

      // Pack probabilities in-lane into the P A-operand (16x32, M=q, K=kpos).
      union { bf16x16 v; bf16 e[16]; } pa;
#pragma unroll
      for (int i = 0; i < 16; ++i) pa.e[i] = (bf16)pv[i];

      // PV accumulate, reusing the shared V B-operands.
#pragma unroll
      for (int dt = 0; dt < 4; ++dt)
        o[j][dt] = wmma_bf16(pa.v, vB[dt], o[j][dt]);
    }
  }

  // Normalize and scatter into ctx[M_ROWS, EMB] (row = s*B+b, col = h*64+d).
  const int d = lane & 15;
  const int b = n / HEADS, h = n % HEADS;
#pragma unroll
  for (int j = 0; j < QBLK; ++j) {
    const float linv = 1.0f / lrun[j];             // lanes q and q+16 agree
#pragma unroll
    for (int r = 0; r < 8; ++r) {
      const float lr = __shfl(linv, r + ((lane & 16) ? 8 : 0), 32);
      const int s = (qt0 + j) * 16 + r + ((lane & 16) ? 8 : 0);
      bf16* dst = ctx + ((size_t)s * BATCH + b) * EMB + h * HD;
      dst[d]      = (bf16)(o[j][0][r] * lr);
      dst[16 + d] = (bf16)(o[j][1][r] * lr);
      dst[32 + d] = (bf16)(o[j][2][r] * lr);
      dst[48 + d] = (bf16)(o[j][3][r] * lr);
    }
  }
}

// ---------------- Stage 4: output projection ------------------------------
// 64x64 output block per wave: per k-step 4 A + 4 B operand loads, 16 WMMAs.
#define OPM 4
#define OPN 4
__global__ __launch_bounds__(256)
void out_proj_kernel(const bf16* __restrict__ ctx,
                     const bf16* __restrict__ wob,
                     const float* __restrict__ bo,
                     float* __restrict__ out) {
  const int lane = threadIdx.x & 31;
  const int wave = threadIdx.x >> 5;
  const int tile = blockIdx.x * 8 + wave;     // (M/64)*(E/64) = 1024 tiles
  const int mt = tile / (EMB / (16 * OPN));
  const int nt = tile % (EMB / (16 * OPN));
  const int m0 = mt * (16 * OPM), n0 = nt * (16 * OPN);

  f32x8 c[OPM][OPN] = {};

  for (int k = 0; k < EMB; k += 32) {
    bf16x16 a[OPM], b[OPN];
#pragma unroll
    for (int i = 0; i < OPM; ++i)
      a[i] = load_a_tile(ctx, EMB, m0 + 16 * i, k, lane);
#pragma unroll
    for (int j = 0; j < OPN; ++j)
      b[j] = load_b_tile(wob, EMB, n0 + 16 * j, k, lane);
#pragma unroll
    for (int i = 0; i < OPM; ++i)
#pragma unroll
      for (int j = 0; j < OPN; ++j)
        c[i][j] = wmma_bf16(a[i], b[j], c[i][j]);
  }

#pragma unroll
  for (int j = 0; j < OPN; ++j) {
    const int ncol = n0 + 16 * j + (lane & 15);
    const float bias = bo[ncol];
#pragma unroll
    for (int i = 0; i < OPM; ++i) {
#pragma unroll
      for (int r = 0; r < 8; ++r) {
        const int m = m0 + 16 * i + r + ((lane & 16) ? 8 : 0);
        out[(size_t)m * EMB + ncol] = c[i][j][r] + bias;
      }
    }
  }
}

// ---------------- Host-side launch ----------------------------------------

extern "C" void kernel_launch(void* const* d_in, const int* in_sizes, int n_in,
                              void* d_out, int out_size, void* d_ws, size_t ws_size,
                              hipStream_t stream) {
  (void)in_sizes; (void)n_in; (void)out_size; (void)ws_size;

  const float* query = (const float*)d_in[0];
  const float* wq    = (const float*)d_in[1];
  const float* bq    = (const float*)d_in[2];
  const float* wk    = (const float*)d_in[3];
  const float* bk    = (const float*)d_in[4];
  const float* wv    = (const float*)d_in[5];
  const float* bv    = (const float*)d_in[6];
  const float* wo    = (const float*)d_in[7];
  const float* bo    = (const float*)d_in[8];
  float* out = (float*)d_out;

  // Workspace layout (bf16 tensors), total ~48 MB.
  char* ws = (char*)d_ws;
  const size_t szX  = (size_t)M_ROWS * EMB * sizeof(bf16);        // 8 MB
  const size_t szW  = (size_t)EMB * EMB * sizeof(bf16);           // 2 MB
  const size_t szQK = (size_t)BHEADS * S_LEN * HD * sizeof(bf16); // 8 MB

  bf16* xb  = (bf16*)(ws);                 ws += szX;
  bf16* wqb = (bf16*)(ws);                 ws += szW;
  bf16* wkb = (bf16*)(ws);                 ws += szW;
  bf16* wvb = (bf16*)(ws);                 ws += szW;
  bf16* wob = (bf16*)(ws);                 ws += szW;
  bf16* qh  = (bf16*)(ws);                 ws += szQK;
  bf16* kh  = (bf16*)(ws);                 ws += szQK;
  bf16* vT  = (bf16*)(ws);                 ws += szQK;
  bf16* ctx = (bf16*)(ws);

  // Stage 1: casts
  {
    const int nX = M_ROWS * EMB;           // 4,194,304
    const int nW = EMB * EMB;              // 1,048,576
    cast_f32_to_bf16<<<(nX + 255) / 256, 256, 0, stream>>>(query, xb, nX);
    cast_f32_to_bf16<<<(nW + 255) / 256, 256, 0, stream>>>(wq, wqb, nW);
    cast_f32_to_bf16<<<(nW + 255) / 256, 256, 0, stream>>>(wk, wkb, nW);
    cast_f32_to_bf16<<<(nW + 255) / 256, 256, 0, stream>>>(wv, wvb, nW);
    cast_f32_to_bf16<<<(nW + 255) / 256, 256, 0, stream>>>(wo, wob, nW);
  }

  // Stage 2: fused QKV projection. (M/64)*(E/16) = 4096 tiles, 8 waves/block.
  qkv_proj_kernel<<<4096 / 8, 256, 0, stream>>>(xb, wqb, wkb, wvb,
                                                bq, bk, bv, qh, kh, vT);

  // Stage 3: flash attention. 32 heads x (128/QBLK) q-tiles, 8 waves/block.
  flash_attn_kernel<<<dim3(BHEADS, (S_LEN / 16) / (8 * QBLK)), 256, 0, stream>>>(
      qh, kh, vT, ctx);

  // Stage 4: output projection. (M/64)*(E/64) = 1024 tiles, 8 waves/block.
  out_proj_kernel<<<1024 / 8, 256, 0, stream>>>(ctx, wob, bo, out);
}
